// Attention_32409823215713
// MI455X (gfx1250) — compile-verified, matrix operand order checked by reading, not run on hardware
//
#include <hip/hip_runtime.h>

// ---------------------------------------------------------------------------
// Types for CDNA5 WMMA (wave32): v_wmma_f32_16x16x32_bf16
// ---------------------------------------------------------------------------
typedef __attribute__((ext_vector_type(16))) __bf16 v16bf;
typedef __attribute__((ext_vector_type(8)))  __bf16 v8bf;
typedef __attribute__((ext_vector_type(8)))  float  v8f;

#define DEV __device__ __forceinline__

// Problem constants (from reference)
constexpr int Bb   = 2;
constexpr int Nseq = 2048;
constexpr int Cc   = 1024;
constexpr int Hh   = 16;
constexpr int Dd   = 64;
constexpr int Mrows = Bb * Nseq;   // 4096
constexpr int Kd    = Cc;          // 1024

// fp32 -> bf16, round-to-nearest-even
DEV __bf16 tobf(float f) {
    unsigned u = __builtin_bit_cast(unsigned, f);
    unsigned r = u + 0x7FFFu + ((u >> 16) & 1u);
    unsigned short h = (unsigned short)(r >> 16);
    return __builtin_bit_cast(__bf16, h);
}

DEV v8bf cvt8(float4 a, float4 b) {
    v8bf r;
    r[0]=tobf(a.x); r[1]=tobf(a.y); r[2]=tobf(a.z); r[3]=tobf(a.w);
    r[4]=tobf(b.x); r[5]=tobf(b.y); r[6]=tobf(b.z); r[7]=tobf(b.w);
    return r;
}

// Raw v_exp_f32 (no libm range fixup). Args here are always <= 0 (or -inf,
// which correctly yields 0), so the fast path is exact for our use.
DEV float fast_exp2(float x) { return __builtin_amdgcn_exp2f(x); }

// Load one 16x32 bf16 A/B fragment (wave32 layout, ISA 7.12.2):
// rowp = pointer to this lane's row (K contiguous), hi = lane>>4.
//   VGPR0-3: lanes0-15 K=0..7,  lanes16-31 K=8..15   -> rowp[hi*8 .. +8)
//   VGPR4-7: lanes0-15 K=16..23,lanes16-31 K=24..31  -> rowp[16+hi*8 .. +8)
// For A the lane indexes the M row; for B the lane indexes the N column with
// K contiguous per column (column-major storage).
DEV v16bf load_frag(const __bf16* rowp, int hi) {
    v8bf lo  = *(const v8bf*)(rowp + hi * 8);
    v8bf hi8 = *(const v8bf*)(rowp + 16 + hi * 8);
    return __builtin_shufflevector(lo, hi8, 0,1,2,3,4,5,6,7,8,9,10,11,12,13,14,15);
}

DEV v8f wmma_bf16(v16bf a, v16bf b, v8f c) {
    // 8 args: (neg_a, A, neg_b, B, c_mod, C, reuse_a, reuse_b)
    return __builtin_amdgcn_wmma_f32_16x16x32_bf16(false, a, false, b, (short)0, c, false, false);
}

// ---------------------------------------------------------------------------
// GEMM: [4096 x Kd] (A) x [Kd x NC] (W, fp32) with bf16 WMMA, fp32 accum.
// MODE 0: A = x (fp32, converted), NC = 3C; epilogue scatters Q(scaled)/K as
//         [B,H,N,D] bf16 and V transposed as [B,H,D,N] bf16 (packed b128).
// MODE 1: A = ctx (bf16), NC = C; epilogue adds bias, writes fp32 out.
// Block tile 128x128, K-step 32, 8 waves in 2(M) x 4(N) grid;
// each wave: 4x2 = 8 accumulator tiles of 16x16.
// ---------------------------------------------------------------------------
template <int MODE>
__global__ __launch_bounds__(256) void
gemm_bf16_kernel(const float* __restrict__ Af, const __bf16* __restrict__ Ab,
                 const float* __restrict__ W, const float* __restrict__ bias,
                 __bf16* __restrict__ qo, __bf16* __restrict__ ko,
                 __bf16* __restrict__ vto, float* __restrict__ out)
{
    constexpr int NC = (MODE == 0) ? 3 * Cc : Cc;
    constexpr int LDT = 40;                     // padded K-stride (bf16) -> 80B rows
    __shared__ __bf16 As[128][LDT];             // [m][k], k contiguous
    __shared__ __bf16 Bs[128][LDT];             // [n][k], k contiguous (column-major W tile)

    const int t    = threadIdx.x;
    const int lane = t & 31;
    const int w    = t >> 5;
    const int wm   = w >> 2;                    // 0..1
    const int wn   = w & 3;                     // 0..3
    const int m0   = blockIdx.y * 128;
    const int n0   = blockIdx.x * 128;
    const int lr   = lane & 15;
    const int hi   = lane >> 4;

    v8f acc[4][2] = {};

    // global->LDS assignments
    const int am  = t >> 1;                     // 0..127 rows of A tile
    const int akc = (t & 1) * 16;               // k chunk 0/16
    const int bk  = t & 31;                     // 0..31 rows (k) of W tile
    const int bnc = (t >> 5) * 16;              // n chunk

    for (int k0 = 0; k0 < Kd; k0 += 32) {
        // ---- A tile ----
        if (MODE == 0) {
            const float4* src = (const float4*)(Af + (size_t)(m0 + am) * Kd + k0 + akc);
            float4 f0 = src[0], f1 = src[1], f2 = src[2], f3 = src[3];
            *(v8bf*)&As[am][akc]     = cvt8(f0, f1);
            *(v8bf*)&As[am][akc + 8] = cvt8(f2, f3);
        } else {
            const v8bf* src = (const v8bf*)(Ab + (size_t)(m0 + am) * Kd + k0 + akc);
            *(v8bf*)&As[am][akc]     = src[0];
            *(v8bf*)&As[am][akc + 8] = src[1];
        }
        // ---- B tile (transpose fp32 row-major -> bf16 column-major) ----
        {
            const float4* src = (const float4*)(W + (size_t)(k0 + bk) * NC + n0 + bnc);
            float4 f[4] = { src[0], src[1], src[2], src[3] };
#pragma unroll
            for (int i = 0; i < 4; ++i) {
                Bs[bnc + i * 4 + 0][bk] = tobf(f[i].x);
                Bs[bnc + i * 4 + 1][bk] = tobf(f[i].y);
                Bs[bnc + i * 4 + 2][bk] = tobf(f[i].z);
                Bs[bnc + i * 4 + 3][bk] = tobf(f[i].w);
            }
        }
        __syncthreads();

        v16bf afr[4], bfr[2];
#pragma unroll
        for (int mi = 0; mi < 4; ++mi)
            afr[mi] = load_frag(&As[wm * 64 + mi * 16 + lr][0], hi);
#pragma unroll
        for (int nj = 0; nj < 2; ++nj)
            bfr[nj] = load_frag(&Bs[wn * 32 + nj * 16 + lr][0], hi);
#pragma unroll
        for (int mi = 0; mi < 4; ++mi)
#pragma unroll
            for (int nj = 0; nj < 2; ++nj)
                acc[mi][nj] = wmma_bf16(afr[mi], bfr[nj], acc[mi][nj]);
        __syncthreads();
    }

    // ---- epilogue (C/D layout: lanes0-15 row=i, lanes16-31 row=i+8; col=lane&15)
#pragma unroll
    for (int mi = 0; mi < 4; ++mi) {
#pragma unroll
        for (int nj = 0; nj < 2; ++nj) {
            if (MODE == 0) {
                const int jbase = n0 + wn * 32 + nj * 16;       // 16-aligned
                const int th    = jbase >> 10;                  // 0=q 1=k 2=v (uniform per tile)
                const int hh    = ((jbase + lr) >> 6) & (Hh - 1);
                const int d     = (jbase + lr) & (Dd - 1);
                const int rbase = m0 + wm * 64 + mi * 16 + hi * 8;
                const int bb    = rbase >> 11;                  // /Nseq
                const int ns    = rbase & (Nseq - 1);
                if (th == 2) {
                    // V transposed [B,H,D,N]: i -> consecutive ns -> one b128 store
                    v8bf pk;
#pragma unroll
                    for (int i = 0; i < 8; ++i) pk[i] = tobf(acc[mi][nj][i]);
                    *(v8bf*)(vto + ((size_t)(bb * Hh + hh) * Dd + d) * Nseq + ns) = pk;
                } else {
                    __bf16* dst = (th == 0) ? qo : ko;
                    const float sc = (th == 0) ? 0.125f : 1.0f; // D^-0.5 folded into Q
#pragma unroll
                    for (int i = 0; i < 8; ++i)
                        dst[((size_t)(bb * Hh + hh) * Nseq + ns + i) * Dd + d] =
                            tobf(acc[mi][nj][i] * sc);
                }
            } else {
#pragma unroll
                for (int i = 0; i < 8; ++i) {
                    const int r = m0 + wm * 64 + mi * 16 + i + hi * 8;
                    const int j = n0 + wn * 32 + nj * 16 + lr;
                    out[(size_t)r * Cc + j] = acc[mi][nj][i] + bias[j];
                }
            }
        }
    }
}

// ---------------------------------------------------------------------------
// Flash attention (transposed-S formulation): grid (N/128, H, B),
// 256 threads = 8 independent waves, each owning 16 query columns.
// Per 32-key block:
//   S^T = K(32xD) . Q^T(Dx16)      -> 4 WMMA (two key tiles x two 32-d chunks)
//   C-layout of S^T: lane = query, VGPRs = keys -> softmax reductions are
//   in-lane trees + ONE shfl_xor(16) each.
//   The two S^T accumulator tiles ARE the B-fragment layout of P^T, so
//   O^T += V^T . P^T needs only an in-register bf16 pack: 4 WMMA, zero LDS.
// ---------------------------------------------------------------------------
__global__ __launch_bounds__(256) void
flash_attn_kernel(const __bf16* __restrict__ Q, const __bf16* __restrict__ K,
                  const __bf16* __restrict__ VT, __bf16* __restrict__ ctx)
{
    const int lane = threadIdx.x & 31;
    const int w    = threadIdx.x >> 5;
    const int lr   = lane & 15;
    const int hi   = lane >> 4;
    const int b    = blockIdx.z;
    const int h    = blockIdx.y;
    const int q0   = blockIdx.x * 128 + w * 16;

    const __bf16* Qb = Q  + ((size_t)(b * Hh + h)) * Nseq * Dd;
    const __bf16* Kb = K  + ((size_t)(b * Hh + h)) * Nseq * Dd;
    const __bf16* Vb = VT + ((size_t)(b * Hh + h)) * Dd * Nseq;

    // Q^T B-operand fragments (lane = query column), resident all kernel
    const v16bf qf0 = load_frag(Qb + (size_t)(q0 + lr) * Dd + 0,  hi);
    const v16bf qf1 = load_frag(Qb + (size_t)(q0 + lr) * Dd + 32, hi);

    v8f oT[4] = {};                 // O^T: 64 d-rows x 16 queries (lane = query)
    float m = -3.0e38f, l = 0.0f;   // one softmax state per lane (per query-half)

    constexpr float L2E = 1.44269504088896f;

    for (int kb = 0; kb < Nseq; kb += 32) {
        // S^T tiles: A = K rows (lane = key), B = Q^T
        v16bf k00 = load_frag(Kb + (size_t)(kb + lr) * Dd + 0,       hi);
        v16bf k01 = load_frag(Kb + (size_t)(kb + lr) * Dd + 32,      hi);
        v16bf k10 = load_frag(Kb + (size_t)(kb + 16 + lr) * Dd + 0,  hi);
        v16bf k11 = load_frag(Kb + (size_t)(kb + 16 + lr) * Dd + 32, hi);
        v8f sT0 = {}, sT1 = {};
        sT0 = wmma_bf16(k00, qf0, sT0);
        sT0 = wmma_bf16(k01, qf1, sT0);
        sT1 = wmma_bf16(k10, qf0, sT1);
        sT1 = wmma_bf16(k11, qf1, sT1);

        // in-lane max over this lane's 16 key scores, combine lane halves
        float rm = fmaxf(sT0[0], sT1[0]);
#pragma unroll
        for (int e = 1; e < 8; ++e) rm = fmaxf(rm, fmaxf(sT0[e], sT1[e]));
        rm = fmaxf(rm, __shfl_xor(rm, 16, 32));

        const float mn    = fmaxf(m, rm);
        const float alpha = fast_exp2((m - mn) * L2E);
        m = mn;

        float p0[8], p1[8], rs = 0.0f;
#pragma unroll
        for (int e = 0; e < 8; ++e) {
            p0[e] = fast_exp2((sT0[e] - mn) * L2E);
            p1[e] = fast_exp2((sT1[e] - mn) * L2E);
            rs += p0[e] + p1[e];
        }
        rs = rs + __shfl_xor(rs, 16, 32);
        l  = l * alpha + rs;

        // P^T B-fragment: elements 0..7 = tile0 VGPRs, 8..15 = tile1 VGPRs
        v16bf pf;
#pragma unroll
        for (int e = 0; e < 8; ++e) { pf[e] = tobf(p0[e]); pf[8 + e] = tobf(p1[e]); }

        // O^T = alpha*O^T + V^T . P^T  (A = V^T rows, keys contiguous)
#pragma unroll
        for (int c = 0; c < 4; ++c) {
            v16bf vf = load_frag(Vb + (size_t)(16 * c + lr) * Nseq + kb, hi);
            oT[c] *= alpha;
            oT[c]  = wmma_bf16(vf, pf, oT[c]);
        }
    }

    // normalize; O^T C-layout: lane = query (lr), vgpr i -> d = 16c + 8*hi + i
    // -> d-contiguous per lane: 4 packed b128 stores into ctx [B,N,C]
    const float inv = 1.0f / l;
    const int   row = q0 + lr;
#pragma unroll
    for (int c = 0; c < 4; ++c) {
        v8bf pk;
#pragma unroll
        for (int i = 0; i < 8; ++i) pk[i] = tobf(oT[c][i] * inv);
        *(v8bf*)(ctx + ((size_t)b * Nseq + row) * Cc + h * Dd + 16 * c + 8 * hi) = pk;
    }
}

// ---------------------------------------------------------------------------
// Host launcher
// ---------------------------------------------------------------------------
extern "C" void kernel_launch(void* const* d_in, const int* in_sizes, int n_in,
                              void* d_out, int out_size, void* d_ws, size_t ws_size,
                              hipStream_t stream)
{
    const float* x      = (const float*)d_in[0];   // [B,N,C]
    const float* w_qkv  = (const float*)d_in[1];   // [C,3C]
    const float* w_proj = (const float*)d_in[2];   // [C,C]
    const float* b_proj = (const float*)d_in[3];   // [C]
    float* out = (float*)d_out;                    // [B,N,C] fp32

    // workspace: 4 x 8 MB bf16 buffers
    const size_t SZ = (size_t)Bb * Hh * Nseq * Dd; // 4,194,304 elems
    char* ws = (char*)d_ws;
    __bf16* q   = (__bf16*)(ws);
    __bf16* k   = (__bf16*)(ws + 2 * SZ);
    __bf16* vT  = (__bf16*)(ws + 4 * SZ);
    __bf16* ctx = (__bf16*)(ws + 6 * SZ);

    // 1) QKV GEMM: [4096x1024] x [1024x3072] -> scatter Q/K/V^T bf16
    gemm_bf16_kernel<0><<<dim3(3 * Cc / 128, Mrows / 128), 256, 0, stream>>>(
        x, nullptr, w_qkv, nullptr, q, k, vT, nullptr);

    // 2) flash attention -> ctx [B,N,C] bf16
    flash_attn_kernel<<<dim3(Nseq / 128, Hh, Bb), 256, 0, stream>>>(q, k, vT, ctx);

    // 3) projection: [4096x1024] x [1024x1024] + bias -> fp32 out
    gemm_bf16_kernel<1><<<dim3(Cc / 128, Mrows / 128), 256, 0, stream>>>(
        nullptr, ctx, w_proj, b_proj, nullptr, nullptr, nullptr, out);
}
